// LSTM_54992761258486
// MI455X (gfx1250) — matrix operand order repaired and checked by reading, not compiled, not run on hardware
//
#include <hip/hip_runtime.h>
#include <hip/hip_bf16.h>

// ---------------------------------------------------------------------------
// CDNA5 WMMA bf16: D(f32 16x16) = A(bf16 16x32) * B(bf16 32x16) + C
// ---------------------------------------------------------------------------
typedef __attribute__((ext_vector_type(16))) __bf16 v16bf;
typedef __attribute__((ext_vector_type(8)))  float  v8f;

__device__ __forceinline__ v8f wmma_bf16(v16bf a, v16bf b, v8f c) {
    // (neg_a, A, neg_b, B, c_mod, C, reuse_a, reuse_b)
    return __builtin_amdgcn_wmma_f32_16x16x32_bf16(false, a, false, b,
                                                   (short)0, c, false, false);
}

__device__ __forceinline__ float sigmoidf_(float x) {
    return 1.0f / (1.0f + __expf(-x));
}

// ---------------------------------------------------------------------------
// Pack W (N x K, row-major fp32) into bf16 B-fragment tiles for Wt (K x N).
// Tile (tn, tk): n in [tn*16,+16), k in [tk*32,+32).
// Per ISA 7.12.2: lane = 16*h + n ; element e -> k = 8*h + e (e<8), 16 + 8*h + (e-8) (e>=8).
// Storage: out[((tn*KT + tk)*32 + lane)*16 + e]  (32B contiguous per lane).
// ---------------------------------------------------------------------------
__global__ void pack_wT_bf16(const float* __restrict__ W, __bf16* __restrict__ out,
                             int N, int K) {
    int KT   = K >> 5;
    int tile = blockIdx.x;            // tn*KT + tk
    int tn   = tile / KT;
    int tk   = tile - tn * KT;
    int lane = threadIdx.x;           // 0..31
    int n     = tn * 16 + (lane & 15);
    int kbase = tk * 32 + (lane >> 4) * 8;
    __bf16* dst      = out + ((size_t)tile * 32 + lane) * 16;
    const float* src = W + (size_t)n * K + kbase;
#pragma unroll
    for (int e = 0; e < 16; ++e) dst[e] = (__bf16)src[e < 8 ? e : e + 8];
}

// ---------------------------------------------------------------------------
// xg = X @ Wt, stored in WMMA C-fragment-tiled layout:
//   xg[(((t*4 + mTile)*128 + nTile)*32 + lane)*8 + r]
// so the recurrence reads its accumulator init as contiguous 16B per lane.
// Block: 256 threads (8 waves). Wave w: M-tile (w&1), N-chunk (w>>1)*64.
// Grid: (M/32, 2048/256). M = T*64 rows; 64 rows/timestep = 4 m-tiles.
// ---------------------------------------------------------------------------
template <bool A_IS_F32>
__global__ void xg_gemm(const void* __restrict__ Xv, const __bf16* __restrict__ Wp,
                        __bf16* __restrict__ xg, int K) {
    const int lane = threadIdx.x & 31;
    const int w    = threadIdx.x >> 5;
    const int m0   = blockIdx.x * 32 + (w & 1) * 16;
    const int n0   = blockIdx.y * 256 + (w >> 1) * 64;
    const int KT   = K >> 5;

    v8f acc[4];
#pragma unroll
    for (int i = 0; i < 4; ++i)
#pragma unroll
        for (int r = 0; r < 8; ++r) acc[i][r] = 0.0f;

    for (int kt = 0; kt < KT; ++kt) {
        v16bf a;
        if (A_IS_F32) {
            const float* X = (const float*)Xv;
            const float* p = X + (size_t)(m0 + (lane & 15)) * K + kt * 32 + (lane >> 4) * 8;
#pragma unroll
            for (int e = 0; e < 8; ++e) { a[e] = (__bf16)p[e]; a[e + 8] = (__bf16)p[e + 16]; }
        } else {
            const __bf16* X = (const __bf16*)Xv;
            const __bf16* p = X + (size_t)(m0 + (lane & 15)) * K + kt * 32 + (lane >> 4) * 8;
#pragma unroll
            for (int e = 0; e < 8; ++e) { a[e] = p[e]; a[e + 8] = p[e + 16]; }
        }
#pragma unroll
        for (int i = 0; i < 4; ++i) {
            int tn = (n0 >> 4) + i;
            v16bf b = *(const v16bf*)(Wp + (((size_t)tn * KT + kt) * 32 + lane) * 16);
            acc[i] = wmma_bf16(a, b, acc[i]);
        }
    }

    // fragment-tiled store: contiguous 16B per lane per tile
    const int t     = m0 >> 6;
    const int mTile = (m0 >> 4) & 3;
#pragma unroll
    for (int i = 0; i < 4; ++i) {
        int nTile = (n0 >> 4) + i;
        __bf16* dst = xg + ((((size_t)t * 4 + mTile) * 128 + nTile) * 32 + lane) * 8;
#pragma unroll
        for (int r = 0; r < 8; ++r) dst[r] = (__bf16)acc[i][r];
    }
}

// ---------------------------------------------------------------------------
// Persistent LSTM recurrence. Grid: 16 blocks x 256 threads = 128 waves.
// Wave wg = bid*8 + widx : mTile = wg&3, j = wg>>2 (0..31).
// Block bid serves j in {bid*2, bid*2+1}; its W_hh tiles (4 gates x 2 j x
// 16 kt = 128 frags = 128 KB) are staged once into dynamic LDS and the
// inner loop reads B fragments with ds_load_b128.
// ---------------------------------------------------------------------------
template <bool WRITE_H1, bool DO_OUT>
__global__ void lstm_rec(const __bf16* __restrict__ xg,     // frag-tiled (see above)
                         const __bf16* __restrict__ Wp,     // packed W_hh^T tiles
                         const float* __restrict__ b_ih, const float* __restrict__ b_hh,
                         float* __restrict__ cbuf,          // (64, 512) f32
                         __bf16* __restrict__ hbuf,         // 2 x (64, 512) bf16
                         __bf16* __restrict__ h1out,        // (T*64, 512) bf16 or null
                         float* __restrict__ out,           // (T*64) f32 or null
                         const float* __restrict__ w_out, const float* __restrict__ b_out,
                         unsigned* __restrict__ bar_cnt, int T) {
    extern __shared__ __bf16 ldsW[];   // 128 frags * 32 lanes * 16 elems = 128 KB
    __shared__ float red[256];

    const int lane  = threadIdx.x & 31;
    const int widx  = threadIdx.x >> 5;
    const int wg    = blockIdx.x * 8 + widx;   // 0..127
    const int mTile = wg & 3;
    const int j     = wg >> 2;                 // 0..31
    const int jl    = widx >> 2;               // 0..1 (local j within block)
    const int m0    = mTile * 16;
    const int n     = lane & 15;
    const int mh    = lane >> 4;
    const int KT    = 16;                      // 512 / 32

    // ---- stage this block's W_hh fragments into LDS (one time) ----
    // frag f = (g*2 + jl)*16 + kt  maps to global tile (g*32 + bid*2 + jl, kt)
    for (int id = threadIdx.x; id < 128 * 64; id += 256) {  // 16B chunks
        int f  = id >> 6, c = id & 63;
        int kt = f & 15, q = f >> 4;          // q = g*2 + jl
        int gjl = q & 1, g = q >> 1;
        size_t tg = (size_t)(g * 32 + blockIdx.x * 2 + gjl) * 16 + kt;
        ((uint4*)ldsW)[id] = ((const uint4*)Wp)[tg * 64 + c];
    }
    __syncthreads();

    float bias[4];
#pragma unroll
    for (int g = 0; g < 4; ++g) {
        int col = g * 512 + j * 16 + n;
        bias[g] = b_ih[col] + b_hh[col];
    }

    for (int t = 0; t < T; ++t) {
        const __bf16* hprev = hbuf + (size_t)(t & 1) * (64 * 512);
        __bf16*       hnew  = hbuf + (size_t)((t + 1) & 1) * (64 * 512);

        // gates init: xg_t (frag-tiled, contiguous per lane) + biases
        v8f acc[4];
#pragma unroll
        for (int g = 0; g < 4; ++g) {
            const __bf16* p = xg +
                ((((size_t)t * 4 + mTile) * 128 + (g * 32 + j)) * 32 + lane) * 8;
#pragma unroll
            for (int r = 0; r < 8; ++r) acc[g][r] = (float)p[r] + bias[g];
        }

        // gates += hprev @ W_hh^T   (K = 512), B frags from LDS
#pragma unroll 2
        for (int kt = 0; kt < KT; ++kt) {
            v16bf a;
            const __bf16* p = hprev + (size_t)(m0 + (lane & 15)) * 512 +
                              kt * 32 + (lane >> 4) * 8;
#pragma unroll
            for (int e = 0; e < 8; ++e) { a[e] = p[e]; a[e + 8] = p[e + 16]; }
#pragma unroll
            for (int g = 0; g < 4; ++g) {
                int f = (g * 2 + jl) * 16 + kt;
                v16bf b = *(const v16bf*)(ldsW + ((size_t)f * 32 + lane) * 16);
                acc[g] = wmma_bf16(a, b, acc[g]);
            }
        }

        // cell update + h write (wave-local)
#pragma unroll
        for (int r = 0; r < 8; ++r) {
            int M = m0 + mh * 8 + r, col = j * 16 + n;
            size_t ci = (size_t)M * 512 + col;
            float ig = sigmoidf_(acc[0][r]);
            float fg = sigmoidf_(acc[1][r]);
            float gg = tanhf(acc[2][r]);
            float og = sigmoidf_(acc[3][r]);
            float cn = fg * cbuf[ci] + ig * gg;
            cbuf[ci] = cn;
            float hv = og * tanhf(cn);
            hnew[ci] = (__bf16)hv;
            if (WRITE_H1)
                h1out[((size_t)t * 64 + M) * 512 + col] = (__bf16)hv;
        }

        // grid-wide barrier (monotonic counter)
        __syncthreads();
        if (threadIdx.x == 0) {
            __threadfence();
            atomicAdd(bar_cnt, 1u);
            unsigned target = (unsigned)(t + 1) * gridDim.x;
            while (__hip_atomic_load(bar_cnt, __ATOMIC_ACQUIRE,
                                     __HIP_MEMORY_SCOPE_AGENT) < target) {
                __builtin_amdgcn_s_sleep(1);
            }
        }
        __syncthreads();

        // fused output head: out[t] = sigmoid(h2(t) . w_out + b_out)
        // hnew stays valid until after barrier(t+1), so this is race-free.
        if (DO_OUT && blockIdx.x == 0) {
            int b = threadIdx.x >> 2, q = threadIdx.x & 3;
            const __bf16* hrow = hnew + (size_t)b * 512 + q * 128;
            const float*  wo   = w_out + q * 128;
            float s = 0.0f;
#pragma unroll 4
            for (int k = 0; k < 128; ++k) s += (float)hrow[k] * wo[k];
            red[threadIdx.x] = s;
            __syncthreads();
            if (q == 0) {
                float tot = red[threadIdx.x] + red[threadIdx.x + 1] +
                            red[threadIdx.x + 2] + red[threadIdx.x + 3] + b_out[0];
                out[(size_t)t * 64 + b] = sigmoidf_(tot);
            }
            __syncthreads();
        }
    }
}

// ---------------------------------------------------------------------------
// Host-side orchestration
// ---------------------------------------------------------------------------
extern "C" void kernel_launch(void* const* d_in, const int* in_sizes, int n_in,
                              void* d_out, int out_size, void* d_ws, size_t ws_size,
                              hipStream_t stream) {
    const int T = 2048, B = 64, IN = 256, H = 512, G = 4 * H; // G = 2048

    const float* input = (const float*)d_in[0];
    const float* W_ih1 = (const float*)d_in[1];
    const float* W_hh1 = (const float*)d_in[2];
    const float* b_ih1 = (const float*)d_in[3];
    const float* b_hh1 = (const float*)d_in[4];
    const float* W_ih2 = (const float*)d_in[5];
    const float* W_hh2 = (const float*)d_in[6];
    const float* b_ih2 = (const float*)d_in[7];
    const float* b_hh2 = (const float*)d_in[8];
    const float* W_out = (const float*)d_in[9];
    const float* b_out = (const float*)d_in[10];
    float* out = (float*)d_out;

    // workspace carve-up (256B aligned)
    char*  base = (char*)d_ws;
    size_t off  = 0;
    auto carve = [&](size_t bytes) {
        char* p = base + off;
        off = (off + bytes + 255) & ~(size_t)255;
        return p;
    };
    __bf16*   xg    = (__bf16*)carve((size_t)T * B * G * 2);      // 512 MiB
    __bf16*   h1buf = (__bf16*)carve((size_t)T * B * H * 2);      // 128 MiB
    __bf16*   pWih1 = (__bf16*)carve((size_t)G * IN * 2);
    __bf16*   pWhh1 = (__bf16*)carve((size_t)G * H * 2);
    __bf16*   pWih2 = (__bf16*)carve((size_t)G * H * 2);
    __bf16*   pWhh2 = (__bf16*)carve((size_t)G * H * 2);
    float*    cbuf  = (float*)carve((size_t)B * H * 4);
    __bf16*   hping = (__bf16*)carve((size_t)2 * B * H * 2);
    unsigned* cnt   = (unsigned*)carve(256);

    const int M = T * B;                 // 131072
    const size_t ldsBytes = 128 * 1024;  // W_hh fragment stage

    // ---- layer 1 ----
    hipMemsetAsync(cnt,   0, 256, stream);
    hipMemsetAsync(cbuf,  0, (size_t)B * H * 4, stream);
    hipMemsetAsync(hping, 0, (size_t)2 * B * H * 2, stream);

    pack_wT_bf16<<<(G / 16) * (IN / 32), 32, 0, stream>>>(W_ih1, pWih1, G, IN);
    pack_wT_bf16<<<(G / 16) * (H  / 32), 32, 0, stream>>>(W_hh1, pWhh1, G, H);
    pack_wT_bf16<<<(G / 16) * (H  / 32), 32, 0, stream>>>(W_ih2, pWih2, G, H);
    pack_wT_bf16<<<(G / 16) * (H  / 32), 32, 0, stream>>>(W_hh2, pWhh2, G, H);

    xg_gemm<true><<<dim3(M / 32, G / 256), 256, 0, stream>>>(input, pWih1, xg, IN);

    lstm_rec<true, false><<<16, 256, ldsBytes, stream>>>(
        xg, pWhh1, b_ih1, b_hh1, cbuf, hping, h1buf,
        nullptr, nullptr, nullptr, cnt + 0, T);

    // ---- layer 2 ----
    hipMemsetAsync(cbuf,  0, (size_t)B * H * 4, stream);
    hipMemsetAsync(hping, 0, (size_t)2 * B * H * 2, stream);

    xg_gemm<false><<<dim3(M / 32, G / 256), 256, 0, stream>>>(h1buf, pWih2, xg, H);

    lstm_rec<false, true><<<16, 256, ldsBytes, stream>>>(
        xg, pWhh2, b_ih2, b_hh2, cbuf, hping, nullptr,
        out, W_out, b_out, cnt + 16, T);

    (void)in_sizes; (void)n_in; (void)out_size; (void)ws_size;
}
